// BridgeNetworkKAN_13082470383758
// MI455X (gfx1250) — matrix operand order repaired and checked by reading, not compiled, not run on hardware
//
#include <hip/hip_runtime.h>
#include <hip/hip_bf16.h>

typedef __attribute__((ext_vector_type(16))) __bf16 v16bf;
typedef __attribute__((ext_vector_type(8)))  __bf16 v8bf;
typedef __attribute__((ext_vector_type(8)))  float  v8f;
typedef __attribute__((ext_vector_type(4)))  float  v4f;

#define BATCH   16384
#define IN_DIM  128
#define HIDDEN  1024
#define STATE_D 64
#define NB      8
#define KDIM1   1152          /* IN_DIM + IN_DIM*NB  (featurized width) */
#define NOUT2   8256          /* 8192 (f_w) + 64 (bias_w) */
#define NPAD2   8448          /* padded to 33*256 so B-frag loads need no guard */

/* ---- workspace layout (bytes) ---- */
#define ACT_OFF  ((size_t)0)                                /* bf16 [BATCH][KDIM1]  */
#define KAN_OFF  (ACT_OFF + (size_t)BATCH*KDIM1*2)          /* bf16 [BATCH][HIDDEN] */
#define WC_OFF   (KAN_OFF + (size_t)BATCH*HIDDEN*2)         /* bf16 [HIDDEN][KDIM1] (N-major, transposed) */
#define GC_OFF   (WC_OFF  + (size_t)HIDDEN*KDIM1*2)         /* bf16 [NPAD2][HIDDEN] (N-major, transposed) */
#define CB_OFF   (GC_OFF  + (size_t)NPAD2*HIDDEN*2)         /* f32  [NPAD2] fused bias vector */

/* ================= fragment loaders (ISA 7.12.2 wave32 layouts) ============ */
/* A 16x32 bf16: lane<16 -> M=lane, K = {0..7} U {16..23}; lanes 16-31 -> +8  */
__device__ inline v16bf load_fragA(const __bf16* base, int ld, int mBase, int k0, int lane) {
  int row = mBase + (lane & 15);
  int kh  = (lane >> 4) * 8;
  const __bf16* p = base + (size_t)row * ld + k0 + kh;
  v16bf f;
  ((v8bf*)&f)[0] = *(const v8bf*)(p);        /* K = k0+kh+0..7   */
  ((v8bf*)&f)[1] = *(const v8bf*)(p + 16);   /* K = k0+kh+16..23 */
  return f;
}
/* B 32x16 bf16 from transposed (N-major) weights: lane n<16 -> col n,
   K = {0..7} U {16..23}; lane n+16 -> +8 (matches A). */
__device__ inline v16bf load_fragB(const __bf16* baseT, int ld, int nBase, int k0, int lane) {
  int n  = nBase + (lane & 15);
  int kh = (lane >> 4) * 8;
  const __bf16* p = baseT + (size_t)n * ld + k0 + kh;
  v16bf f;
  ((v8bf*)&f)[0] = *(const v8bf*)(p);
  ((v8bf*)&f)[1] = *(const v8bf*)(p + 16);
  return f;
}

/* ============ pack W = [scale_base ; scale_sp*coef] transposed to bf16 ====== */
__global__ __launch_bounds__(256) void k_prep_w(const float* __restrict__ coef,
                                                const float* __restrict__ scale_base,
                                                const float* __restrict__ scale_sp,
                                                __bf16* __restrict__ Wc) {
  int idx = blockIdx.x * 256 + threadIdx.x;
  if (idx >= HIDDEN * KDIM1) return;
  int n = idx / KDIM1, k = idx % KDIM1;       /* Wc[n][k] = W[k][n] */
  float v;
  if (k < IN_DIM) {
    v = scale_base[k * HIDDEN + n];
  } else {
    int t = k - IN_DIM, i = t >> 3, c = t & 7;
    v = scale_sp[i * HIDDEN + n] * coef[((size_t)i * HIDDEN + n) * NB + c];
  }
  Wc[idx] = (__bf16)v;
}

/* ======= pack G = [f_w | bias_w] transposed + fused bias vector cb ========= */
__global__ __launch_bounds__(256) void k_prep_g(const float* __restrict__ f_w,
                                                const float* __restrict__ bias_w,
                                                const float* __restrict__ f_b,
                                                const float* __restrict__ bias_b,
                                                __bf16* __restrict__ Gc,
                                                float* __restrict__ cb) {
  int idx = blockIdx.x * 256 + threadIdx.x;
  if (idx >= NPAD2 * HIDDEN) return;
  int n = idx / HIDDEN, k = idx % HIDDEN;     /* Gc[n][k] */
  float v = 0.0f;
  if (n < 8192)       v = f_w[(size_t)k * 8192 + n];
  else if (n < NOUT2) v = bias_w[k * STATE_D + (n - 8192)];
  Gc[idx] = (__bf16)v;
  if (k == 0) {
    float c = 0.0f;
    if (n < 8192)       c = f_b[n];
    else if (n < NOUT2) c = bias_b[n - 8192];
    cb[n] = c;
  }
}

/* ========== featurize: act[b] = [ silu(x) (128) | bspline bases (1024) ] === */
__global__ __launch_bounds__(256) void k_featurize(const float* __restrict__ state,
                                                   const float* __restrict__ novelU,
                                                   __bf16* __restrict__ act) {
  int idx = blockIdx.x * 256 + threadIdx.x;   /* b*128 + i */
  if (idx >= BATCH * IN_DIM) return;
  int b = idx >> 7, i = idx & 127;
  float x = (i < STATE_D) ? state[b * STATE_D + i] : novelU[b * STATE_D + (i - STATE_D)];
  float s = x / (1.0f + __expf(-x));          /* silu */
  /* Cox-de-Boor cubic B-splines on uniform grid h=0.4, knots t_j=-1+0.4*(j-3) */
  const float h = 0.4f;
  float bb[11];
#pragma unroll
  for (int m = 0; m < 11; ++m) {
    float t0 = -1.0f + h * (float)(m - 3);
    bb[m] = (x >= t0 && x < t0 + h) ? 1.0f : 0.0f;
  }
#pragma unroll
  for (int j = 1; j <= 3; ++j) {
    float inv = 1.0f / (h * (float)j);
#pragma unroll
    for (int m = 0; m <= 10 - j; ++m) {
      float tm   = -1.0f + h * (float)(m - 3);
      float tmj1 = -1.0f + h * (float)(m + j + 1 - 3);
      bb[m] = (x - tm) * inv * bb[m] + (tmj1 - x) * inv * bb[m + 1];
    }
  }
  __bf16* row = act + (size_t)b * KDIM1;
  row[i] = (__bf16)s;
  v8bf pk;
#pragma unroll
  for (int c = 0; c < 8; ++c) pk[c] = (__bf16)bb[c];
  *(v8bf*)(row + IN_DIM + i * 8) = pk;        /* 16B aligned: 2*(128+8i), 2304|b */
}

/* ============== GEMM1: kan[B,1024] = act[B,1152] @ W  (bf16 out) =========== */
__global__ __launch_bounds__(256) void k_gemm_kan(const __bf16* __restrict__ act,
                                                  const __bf16* __restrict__ Wc,
                                                  __bf16* __restrict__ kan) {
  int lane = threadIdx.x & 31;
  int wave = threadIdx.x >> 5;
  int wm = wave & 1, wn = wave >> 1;
  int mBase = blockIdx.y * 64 + wm * 32;      /* wave tile: 32 x 64 */
  int nBase = blockIdx.x * 256 + wn * 64;
  v8f acc[2][4] = {};
  for (int k0 = 0; k0 < KDIM1; k0 += 32) {
    v16bf a0 = load_fragA(act, KDIM1, mBase,      k0, lane);
    v16bf a1 = load_fragA(act, KDIM1, mBase + 16, k0, lane);
#pragma unroll
    for (int j = 0; j < 4; ++j) {
      v16bf bf = load_fragB(Wc, KDIM1, nBase + 16 * j, k0, lane);
      acc[0][j] = __builtin_amdgcn_wmma_f32_16x16x32_bf16(false, a0, false, bf, (short)0, acc[0][j], false, false);
      acc[1][j] = __builtin_amdgcn_wmma_f32_16x16x32_bf16(false, a1, false, bf, (short)0, acc[1][j], false, false);
    }
  }
  int col = lane & 15, rofs = (lane >> 4) * 8;
#pragma unroll
  for (int mi = 0; mi < 2; ++mi)
#pragma unroll
    for (int j = 0; j < 4; ++j)
#pragma unroll
      for (int v = 0; v < 8; ++v)
        kan[(size_t)(mBase + mi * 16 + rofs + v) * HIDDEN + nBase + 16 * j + col] =
            (__bf16)acc[mi][j][v];
}

/* ==== GEMM2: [B,1024] @ [1024,8256] -> AB (fp32, +f_b) and bias (+bias_b) == */
__global__ __launch_bounds__(256) void k_gemm_out(const __bf16* __restrict__ kan,
                                                  const __bf16* __restrict__ Gc,
                                                  const float* __restrict__ cb,
                                                  float* __restrict__ AB,
                                                  float* __restrict__ biasOut) {
  int lane = threadIdx.x & 31;
  int wave = threadIdx.x >> 5;
  int wm = wave & 1, wn = wave >> 1;
  int mBase = blockIdx.y * 64 + wm * 32;
  int nBase = blockIdx.x * 256 + wn * 64;
  v8f acc[2][4] = {};
  for (int k0 = 0; k0 < HIDDEN; k0 += 32) {
    v16bf a0 = load_fragA(kan, HIDDEN, mBase,      k0, lane);
    v16bf a1 = load_fragA(kan, HIDDEN, mBase + 16, k0, lane);
#pragma unroll
    for (int j = 0; j < 4; ++j) {
      v16bf bf = load_fragB(Gc, HIDDEN, nBase + 16 * j, k0, lane);
      acc[0][j] = __builtin_amdgcn_wmma_f32_16x16x32_bf16(false, a0, false, bf, (short)0, acc[0][j], false, false);
      acc[1][j] = __builtin_amdgcn_wmma_f32_16x16x32_bf16(false, a1, false, bf, (short)0, acc[1][j], false, false);
    }
  }
  int col = lane & 15, rofs = (lane >> 4) * 8;
#pragma unroll
  for (int j = 0; j < 4; ++j) {
    int n = nBase + 16 * j + col;
    float add = cb[n];
#pragma unroll
    for (int mi = 0; mi < 2; ++mi)
#pragma unroll
      for (int v = 0; v < 8; ++v) {
        int m = mBase + mi * 16 + rofs + v;
        float val = acc[mi][j][v] + add;
        if (n < 8192)       AB[(size_t)m * 8192 + n] = val;
        else if (n < NOUT2) biasOut[(size_t)m * STATE_D + (n - 8192)] = val;
      }
  }
}

/* ====== out[b,s] = dot(AB[b,s,:], x[b,:]) + bias[b,s]  (one wave per row) == */
__global__ __launch_bounds__(256) void k_bmm(const float* __restrict__ AB,
                                             const float* __restrict__ biasOut,
                                             const float* __restrict__ state,
                                             const float* __restrict__ novelU,
                                             float* __restrict__ out) {
  int lane = threadIdx.x & 31;
  int wave = threadIdx.x >> 5;
  long long gw = (long long)blockIdx.x * 8 + wave;  /* over BATCH*STATE_D rows */
  int b = (int)(gw >> 6), s = (int)(gw & 63);
  int u0 = lane * 4;                                 /* lanes 0-15: state, 16-31: novelU */
  const float* xsrc = (u0 < STATE_D) ? (state + (size_t)b * STATE_D + u0)
                                     : (novelU + (size_t)b * STATE_D + (u0 - STATE_D));
  v4f xv = *(const v4f*)xsrc;
  v4f av = *(const v4f*)(AB + ((size_t)b * STATE_D + s) * IN_DIM + u0);
  float p = xv[0] * av[0] + xv[1] * av[1] + xv[2] * av[2] + xv[3] * av[3];
#pragma unroll
  for (int o = 16; o > 0; o >>= 1) p += __shfl_xor(p, o, 32);
  if (lane == 0) out[(size_t)b * STATE_D + s] = p + biasOut[(size_t)b * STATE_D + s];
}

extern "C" void kernel_launch(void* const* d_in, const int* in_sizes, int n_in,
                              void* d_out, int out_size, void* d_ws, size_t ws_size,
                              hipStream_t stream) {
  const float* novelU     = (const float*)d_in[0];
  const float* state      = (const float*)d_in[1];
  const float* coef       = (const float*)d_in[2];
  const float* scale_base = (const float*)d_in[3];
  const float* scale_sp   = (const float*)d_in[4];
  const float* bias_w     = (const float*)d_in[5];
  const float* bias_b     = (const float*)d_in[6];
  const float* f_w        = (const float*)d_in[7];
  const float* f_b        = (const float*)d_in[8];

  float* out     = (float*)d_out;                          /* [B,64]        */
  float* AB      = out + (size_t)BATCH * STATE_D;          /* [B,64,128]    */
  float* biasOut = AB + (size_t)BATCH * 8192;              /* [B,64]        */

  char*   ws  = (char*)d_ws;
  __bf16* act = (__bf16*)(ws + ACT_OFF);
  __bf16* kan = (__bf16*)(ws + KAN_OFF);
  __bf16* Wc  = (__bf16*)(ws + WC_OFF);
  __bf16* Gc  = (__bf16*)(ws + GC_OFF);
  float*  cb  = (float*)(ws + CB_OFF);

  k_prep_w<<<(HIDDEN * KDIM1 + 255) / 256, 256, 0, stream>>>(coef, scale_base, scale_sp, Wc);
  k_prep_g<<<(NPAD2 * HIDDEN + 255) / 256, 256, 0, stream>>>(f_w, bias_w, f_b, bias_b, Gc, cb);
  k_featurize<<<(BATCH * IN_DIM + 255) / 256, 256, 0, stream>>>(state, novelU, act);

  /* GEMM1: M=16384 N=1024 K=1152 ; block tile 64x256 */
  k_gemm_kan<<<dim3(HIDDEN / 256, BATCH / 64), 256, 0, stream>>>(act, Wc, kan);
  /* GEMM2: M=16384 N=8448(pad) K=1024 */
  k_gemm_out<<<dim3(NPAD2 / 256, BATCH / 64), 256, 0, stream>>>(kan, Gc, cb, AB, biasOut);
  /* out = AB . x + bias : one wave per (b,s), 8 waves per block */
  k_bmm<<<(BATCH * STATE_D) / 8, 256, 0, stream>>>(AB, biasOut, state, novelU, out);
}
